// Attention_10264971837751
// MI455X (gfx1250) — matrix operand order repaired and checked by reading, not compiled, not run on hardware
//
#include <hip/hip_runtime.h>
#include <hip/hip_bf16.h>

// ---- problem constants (match reference) ----
#define B_   8
#define T_   2048
#define D_   1024
#define HS_  64

typedef __attribute__((ext_vector_type(16))) _Float16 v16h;
typedef __attribute__((ext_vector_type(8)))  _Float16 v8h;
typedef __attribute__((ext_vector_type(8)))  float    v8f;
typedef __attribute__((ext_vector_type(4)))  float    v4f;

__device__ __forceinline__ float rowmax16(float v) {
#pragma unroll
    for (int o = 1; o < 16; o <<= 1) v = fmaxf(v, __shfl_xor(v, o, 32));
    return v;
}
__device__ __forceinline__ float rowsum16(float v) {
#pragma unroll
    for (int o = 1; o < 16; o <<= 1) v += __shfl_xor(v, o, 32);
    return v;
}

// ---------------------------------------------------------------------------
// Pass 1: QKV projection.  grid = (B*T/16, HS/16, 3), block = 32 (one wave).
// Each wave computes a 16x16 tile of q/k/v via WMMA f16->f32 over D=1024.
// q,k stored row-major f16 [B*T][HS]; v stored transposed f16 [B][HS][T].
// ---------------------------------------------------------------------------
__global__ __launch_bounds__(32) void qkv_proj_kernel(
    const float* __restrict__ x,
    const float* __restrict__ wq, const float* __restrict__ bq,
    const float* __restrict__ wk, const float* __restrict__ bk,
    const float* __restrict__ wv, const float* __restrict__ bv,
    _Float16* __restrict__ q16, _Float16* __restrict__ k16,
    _Float16* __restrict__ vT16)
{
    const int lane = threadIdx.x;
    const int half = lane >> 4;   // 0: lanes 0-15, 1: lanes 16-31
    const int l16  = lane & 15;

    const int rowTile = blockIdx.x;   // 16 flat tokens
    const int colTile = blockIdx.y;   // 16 head dims
    const int matSel  = blockIdx.z;   // 0=q 1=k 2=v

    const float* w    = (matSel == 0) ? wq : (matSel == 1) ? wk : wv;
    const float* bias = (matSel == 0) ? bq : (matSel == 1) ? bk : bv;

    const int arowIdx = rowTile * 16 + l16;   // token (A-matrix row, per lane)
    const int bcolIdx = colTile * 16 + l16;   // head dim (B-matrix column, per lane)

    const float* arow = x + (size_t)arowIdx * D_;
    const float* brow = w + (size_t)bcolIdx * D_;

    v8f c = {};
    for (int k = 0; k < D_; k += 32) {
        // A operand in this code's contiguous k order: half0 holds K k+0..15 of its row; half1 K k+16..31
        const int ka = k + half * 16;
        v4f a0 = *(const v4f*)(arow + ka);
        v4f a1 = *(const v4f*)(arow + ka + 4);
        v4f a2 = *(const v4f*)(arow + ka + 8);
        v4f a3 = *(const v4f*)(arow + ka + 12);
        // B operand, same contiguous k order as A: half0 holds K k+0..15 of its column; half1 K k+16..31
        const int kb = k + half * 16;
        v4f b0 = *(const v4f*)(brow + kb);
        v4f b1 = *(const v4f*)(brow + kb + 4);
        v4f b2 = *(const v4f*)(brow + kb + 8);
        v4f b3 = *(const v4f*)(brow + kb + 12);

        v16h a, bb;
#pragma unroll
        for (int i = 0; i < 4; i++) {
            a[i]      = (_Float16)a0[i];
            a[4 + i]  = (_Float16)a1[i];
            a[8 + i]  = (_Float16)a2[i];
            a[12 + i] = (_Float16)a3[i];
            bb[i]      = (_Float16)b0[i];
            bb[4 + i]  = (_Float16)b1[i];
            bb[8 + i]  = (_Float16)b2[i];
            bb[12 + i] = (_Float16)b3[i];
        }
        c = __builtin_amdgcn_wmma_f32_16x16x32_f16(false, a, false, bb,
                                                   (short)0, c, false, false);
    }

    // bias: C/D element (m = jj + half*8, n = l16) -> bias index is n for all jj
    const float bv_ = bias[colTile * 16 + l16];
    const int mBase = rowTile * 16 + half * 8;

    if (matSel < 2) {
        _Float16* o = (matSel == 0) ? q16 : k16;
#pragma unroll
        for (int jj = 0; jj < 8; jj++) {
            o[(size_t)(mBase + jj) * HS_ + colTile * 16 + l16] =
                (_Float16)(c[jj] + bv_);
        }
    } else {
#pragma unroll
        for (int jj = 0; jj < 8; jj++) {
            const int tok = mBase + jj;
            const int b = tok >> 11;          // /T_
            const int t = tok & (T_ - 1);
            vT16[((size_t)b * HS_ + colTile * 16 + l16) * T_ + t] =
                (_Float16)(c[jj] + bv_);
        }
    }
}

// ---------------------------------------------------------------------------
// Pass 2: causal flash attention.  grid = B*T/16, block = 32 (one wave).
// One wave owns 16 query rows; keys processed 32 at a time:
//   S = Q*K^T (4 WMMA), online softmax (shfl reductions), P via LDS transpose,
//   O += P*V (4 WMMA, one per 16-wide head chunk).
// ---------------------------------------------------------------------------
__global__ __launch_bounds__(32) void flash_attn_kernel(
    const _Float16* __restrict__ q16,
    const _Float16* __restrict__ k16,
    const _Float16* __restrict__ vT16,
    float* __restrict__ out)
{
    __shared__ __align__(32) _Float16 pl[16 * 32];   // P tile staging (1 KB)

    const int lane = threadIdx.x;
    const int half = lane >> 4;
    const int l16  = lane & 15;

    const int tile = blockIdx.x;                 // 0 .. B*T/16-1
    const int b    = tile >> 7;                  // T_/16 = 128 tiles per batch
    const int qr   = (tile & 127) * 16;          // first query row in batch

    const _Float16* qb = q16  + (size_t)b * T_ * HS_;
    const _Float16* kbp = k16 + (size_t)b * T_ * HS_;
    const _Float16* vb = vT16 + (size_t)b * HS_ * T_;

    // ---- load Q A-fragments (16 rows x 64 K), split into two K=32 chunks ----
    const _Float16* qrp = qb + (size_t)(qr + l16) * HS_;
    v8h q00 = *(const v8h*)(qrp + half * 16);        // K 0..7  / 16..23
    v8h q01 = *(const v8h*)(qrp + 8 + half * 16);    // K 8..15 / 24..31
    v8h q10 = *(const v8h*)(qrp + 32 + half * 16);   // K 32..39 / 48..55
    v8h q11 = *(const v8h*)(qrp + 40 + half * 16);   // K 40..47 / 56..63
    v16h qa0, qa1;
#pragma unroll
    for (int i = 0; i < 8; i++) {
        qa0[i] = q00[i]; qa0[8 + i] = q01[i];
        qa1[i] = q10[i]; qa1[8 + i] = q11[i];
    }

    float mi[8], li[8];
#pragma unroll
    for (int jj = 0; jj < 8; jj++) { mi[jj] = -1.0e30f; li[jj] = 0.0f; }
    v8f o0 = {}, o1 = {}, o2 = {}, o3 = {};

    const float scale = 0.125f;          // 1/sqrt(64)
    const float L2E   = 1.44269504f;

    for (int kb = 0; kb <= qr; kb += 32) {
        // ---- K B-fragments: two 16-key subtiles, two K-dim chunks each ----
        const _Float16* kr0 = kbp + (size_t)(kb + l16) * HS_;
        const _Float16* kr1 = kbp + (size_t)(kb + 16 + l16) * HS_;
        v16h kf00 = *(const v16h*)(kr0 + half * 16);        // keys kb..+15, K 0..31
        v16h kf01 = *(const v16h*)(kr0 + 32 + half * 16);   // keys kb..+15, K 32..63
        v16h kf10 = *(const v16h*)(kr1 + half * 16);
        v16h kf11 = *(const v16h*)(kr1 + 32 + half * 16);

        v8f s0 = {}, s1 = {};
        s0 = __builtin_amdgcn_wmma_f32_16x16x32_f16(false, qa0, false, kf00, (short)0, s0, false, false);
        s0 = __builtin_amdgcn_wmma_f32_16x16x32_f16(false, qa1, false, kf01, (short)0, s0, false, false);
        s1 = __builtin_amdgcn_wmma_f32_16x16x32_f16(false, qa0, false, kf10, (short)0, s1, false, false);
        s1 = __builtin_amdgcn_wmma_f32_16x16x32_f16(false, qa1, false, kf11, (short)0, s1, false, false);

        // ---- scale + causal mask (only the diagonal block needs masking) ----
#pragma unroll
        for (int jj = 0; jj < 8; jj++) { s0[jj] *= scale; s1[jj] *= scale; }
        if (kb + 31 > qr) {
            const int mrow = qr + half * 8;
#pragma unroll
            for (int jj = 0; jj < 8; jj++) {
                const int m = mrow + jj;
                if (kb + l16 > m)      s0[jj] = -1.0e30f;
                if (kb + 16 + l16 > m) s1[jj] = -1.0e30f;
            }
        }

        // ---- online softmax + stage P into LDS ----
        __syncthreads();
#pragma unroll
        for (int jj = 0; jj < 8; jj++) {
            const float rmax = rowmax16(fmaxf(s0[jj], s1[jj]));
            const float mnew = fmaxf(mi[jj], rmax);
            const float corr = __builtin_exp2f((mi[jj] - mnew) * L2E);
            const float p0 = __builtin_exp2f((s0[jj] - mnew) * L2E);
            const float p1 = __builtin_exp2f((s1[jj] - mnew) * L2E);
            li[jj] = li[jj] * corr + rowsum16(p0 + p1);
            mi[jj] = mnew;
            o0[jj] *= corr; o1[jj] *= corr; o2[jj] *= corr; o3[jj] *= corr;
            pl[(jj + half * 8) * 32 + l16]      = (_Float16)p0;
            pl[(jj + half * 8) * 32 + 16 + l16] = (_Float16)p1;
        }
        __syncthreads();

        // ---- reload P in A-layout (16x32) ----
        const _Float16* pr = &pl[l16 * 32 + half * 16];
        v8h plo = *(const v8h*)(pr);        // K 0..7 / 16..23
        v8h phi = *(const v8h*)(pr + 8);    // K 8..15 / 24..31
        v16h pa;
#pragma unroll
        for (int i = 0; i < 8; i++) { pa[i] = plo[i]; pa[8 + i] = phi[i]; }
        __syncthreads();

        // ---- O += P * V, one WMMA per 16-wide head chunk (V transposed) ----
        const _Float16* vpBase = vb + (size_t)l16 * T_ + kb + half * 16;
        v16h vf0 = *(const v16h*)(vpBase);
        v16h vf1 = *(const v16h*)(vpBase + 16 * T_);
        v16h vf2 = *(const v16h*)(vpBase + 32 * T_);
        v16h vf3 = *(const v16h*)(vpBase + 48 * T_);
        o0 = __builtin_amdgcn_wmma_f32_16x16x32_f16(false, pa, false, vf0, (short)0, o0, false, false);
        o1 = __builtin_amdgcn_wmma_f32_16x16x32_f16(false, pa, false, vf1, (short)0, o1, false, false);
        o2 = __builtin_amdgcn_wmma_f32_16x16x32_f16(false, pa, false, vf2, (short)0, o2, false, false);
        o3 = __builtin_amdgcn_wmma_f32_16x16x32_f16(false, pa, false, vf3, (short)0, o3, false, false);
    }

    // ---- normalize and store [B,T,HS] f32 ----
#pragma unroll
    for (int jj = 0; jj < 8; jj++) {
        const float inv = 1.0f / li[jj];
        const int m = qr + half * 8 + jj;
        float* op = out + ((size_t)b * T_ + m) * HS_;
        op[l16]      = o0[jj] * inv;
        op[16 + l16] = o1[jj] * inv;
        op[32 + l16] = o2[jj] * inv;
        op[48 + l16] = o3[jj] * inv;
    }
}

// ---------------------------------------------------------------------------
extern "C" void kernel_launch(void* const* d_in, const int* in_sizes, int n_in,
                              void* d_out, int out_size, void* d_ws, size_t ws_size,
                              hipStream_t stream) {
    const float* x  = (const float*)d_in[0];
    const float* wq = (const float*)d_in[1];
    const float* bq = (const float*)d_in[2];
    const float* wk = (const float*)d_in[3];
    const float* bk = (const float*)d_in[4];
    const float* wv = (const float*)d_in[5];
    const float* bv = (const float*)d_in[6];

    const size_t nTok = (size_t)B_ * T_;          // 16384
    _Float16* q16  = (_Float16*)d_ws;             // 2 MB
    _Float16* k16  = q16 + nTok * HS_;            // 2 MB
    _Float16* vT16 = k16 + nTok * HS_;            // 2 MB (transposed [B][HS][T])

    dim3 g1((unsigned)(nTok / 16), HS_ / 16, 3);  // (1024, 4, 3)
    qkv_proj_kernel<<<g1, 32, 0, stream>>>(x, wq, bq, wk, bk, wv, bv,
                                           q16, k16, vT16);

    dim3 g2((unsigned)(nTok / 16));               // 1024 query tiles
    flash_attn_kernel<<<g2, 32, 0, stream>>>(q16, k16, vT16, (float*)d_out);
}
